// LogMCDropoutVariance_11553462026300
// MI455X (gfx1250) — matrix operand order repaired and checked by reading, not compiled, not run on hardware
//
#include <hip/hip_runtime.h>
#include <stdint.h>

typedef __attribute__((ext_vector_type(16))) _Float16 v16h;
typedef __attribute__((ext_vector_type(8)))  _Float16 v8h;
typedef __attribute__((ext_vector_type(8)))  float    v8f;

#define S_N        10
#define KEEP_INV   (1.0f / 0.9f)
#define KEEP_THRESH 3865470566u   /* floor(0.9 * 2^32) */
#define BDIM       32768
#define FDIM       512
#define CDIM       1000
#define CPAD       1024
#define PITCH      520            /* halves per LDS feature row (16B-aligned, bank-skewed) */

// ---------------------------------------------------------------------------
// Prep 1: pack W (512x1000 f32) into a contiguous-K v_wmma_f32_16x16x32_f16
// B-fragment layout (not the ISA table order): frag[kstep][ntile][lane][e], where
// lane = 16*h + n, element e -> W row (32*kstep + 16*h + e), col (16*ntile + n).
// Each lane's 16 halves are then contiguous (32B) in the hot loop.
// ---------------------------------------------------------------------------
__global__ void pack_w_frag(const float* __restrict__ W, _Float16* __restrict__ whfrag) {
  const int id    = blockIdx.x * blockDim.x + threadIdx.x;   // 16*64*32*16 = 524288
  const int e     = id & 15;
  const int lane  = (id >> 4) & 31;
  const int ntile = (id >> 9) & 63;
  const int kstep = id >> 15;
  const int h = lane >> 4, n = lane & 15;
  const int row = 32 * kstep + 16 * h + e;                   // < 512 always
  const int col = ntile * 16 + n;                            // < 1024
  const float v = (col < CDIM) ? W[(size_t)row * CDIM + col] : 0.0f;
  whfrag[id] = (_Float16)v;
}

// Prep 2: pad bias to 1024; padded classes get -1e30 so softmax prob == 0.
__global__ void pack_bias(const float* __restrict__ b, float* __restrict__ bpad) {
  const int c = blockIdx.x * blockDim.x + threadIdx.x;
  if (c < CPAD) bpad[c] = (c < CDIM) ? b[c] : -1.0e30f;
}

// Deterministic per-(s,b,f) Bernoulli hash (lowbias32 mix).
__device__ __forceinline__ uint32_t mix32(uint32_t x) {
  x ^= x >> 16; x *= 0x7feb352dU;
  x ^= x >> 15; x *= 0x846ca68bU;
  x ^= x >> 16;
  return x;
}

// Row reduction: 16 rows of the block's M-tile. Phase 1: shfl_xor within the
// 16-lane column groups; phase 2: tiny LDS reduce across the 8 waves.
template <bool MAXOP>
__device__ __forceinline__ void row_reduce(float v[8], float out[8],
                                           int lane, int wv,
                                           float (*partial)[16], float* rowval) {
  const int h = lane >> 4;
#pragma unroll
  for (int r = 0; r < 8; ++r) {
    float x = v[r];
#pragma unroll
    for (int off = 8; off >= 1; off >>= 1) {
      float y = __shfl_xor(x, off, 16);
      x = MAXOP ? fmaxf(x, y) : (x + y);
    }
    v[r] = x;
  }
  if ((lane & 15) == 0) {
#pragma unroll
    for (int r = 0; r < 8; ++r) partial[wv][h * 8 + r] = v[r];
  }
  __syncthreads();
  if (threadIdx.x < 16) {
    float x = partial[0][threadIdx.x];
#pragma unroll
    for (int w = 1; w < 8; ++w) {
      float y = partial[w][threadIdx.x];
      x = MAXOP ? fmaxf(x, y) : (x + y);
    }
    rowval[threadIdx.x] = x;
  }
  __syncthreads();
#pragma unroll
  for (int r = 0; r < 8; ++r) out[r] = rowval[h * 8 + r];
}

// Softmax over the block's 1024 columns for one sample's logit fragments,
// then accumulate probability first/second moments (all in VGPRs).
__device__ __forceinline__ void softmax_accum(v8f* d, const float* bb,
                                              v8f* accS, v8f* accS2,
                                              int lane, int wv,
                                              float (*partial)[16], float* rowval) {
  float pm[8];
#pragma unroll
  for (int r = 0; r < 8; ++r) pm[r] = -3.0e38f;
#pragma unroll
  for (int nt = 0; nt < 8; ++nt) {
#pragma unroll
    for (int r = 0; r < 8; ++r) {
      const float x = d[nt][r] + bb[nt];
      d[nt][r] = x;
      pm[r] = fmaxf(pm[r], x);
    }
  }
  float rmax[8];
  row_reduce<true>(pm, rmax, lane, wv, partial, rowval);

  float ps[8];
#pragma unroll
  for (int r = 0; r < 8; ++r) ps[r] = 0.0f;
#pragma unroll
  for (int nt = 0; nt < 8; ++nt) {
#pragma unroll
    for (int r = 0; r < 8; ++r) {
      const float e = __expf(d[nt][r] - rmax[r]);
      d[nt][r] = e;
      ps[r] += e;
    }
  }
  float rsum[8];
  row_reduce<false>(ps, rsum, lane, wv, partial, rowval);
  float rinv[8];
#pragma unroll
  for (int r = 0; r < 8; ++r) rinv[r] = 1.0f / rsum[r];

#pragma unroll
  for (int nt = 0; nt < 8; ++nt) {
#pragma unroll
    for (int r = 0; r < 8; ++r) {
      const float p = d[nt][r] * rinv[r];
      accS[nt][r]  += p;
      accS2[nt][r] += p * p;
    }
  }
}

// ---------------------------------------------------------------------------
// Fused MC-dropout -> GEMM(WMMA f16) -> softmax -> variance -> log kernel.
// Block: 256 threads (8 wave32). Block owns 16 rows of B; wave wv owns
// columns [wv*128, wv*128+128) = 8 WMMA N-tiles. Two MC samples share each
// B-fragment load (halves L2 W traffic, doubles WMMA ILP). All 8 B fragments
// of a K-step are preloaded into distinct buffers so the scheduler can keep
// them in flight behind the WMMA stream (incremental s_wait_loadcnt).
// ---------------------------------------------------------------------------
__global__ __launch_bounds__(256, 1) void mc_dropout_var(
    const float* __restrict__ feat, const v16h* __restrict__ whfrag,
    const float* __restrict__ bpad, float* __restrict__ out) {
  __shared__ __align__(16) _Float16 featB [16 * PITCH];  // base f16 features
  __shared__ __align__(16) _Float16 featA0[16 * PITCH];  // masked, sample 2p
  __shared__ __align__(16) _Float16 featA1[16 * PITCH];  // masked, sample 2p+1
  __shared__ float partial[8][16];
  __shared__ float rowval[16];

  const int tid  = threadIdx.x;
  const int lane = tid & 31;
  const int wv   = tid >> 5;
  const int h    = lane >> 4;
  const int n    = lane & 15;
  const int m16  = lane & 15;
  const int row0 = blockIdx.x * 16;

  // Load + convert this block's 16x512 feature tile once.
  for (int i = tid; i < 16 * FDIM; i += 256) {
    const int m = i >> 9, k = i & (FDIM - 1);
    featB[m * PITCH + k] = (_Float16)feat[(size_t)(row0 + m) * FDIM + k];
  }

  // Bias for this wave's 128 columns (col fixed per lane within a tile).
  float bb[8];
#pragma unroll
  for (int nt = 0; nt < 8; ++nt) bb[nt] = bpad[wv * 128 + nt * 16 + n];

  v8f accS[8], accS2[8];
#pragma unroll
  for (int nt = 0; nt < 8; ++nt) { accS[nt] = (v8f)0.0f; accS2[nt] = (v8f)0.0f; }

  for (int sp = 0; sp < S_N / 2; ++sp) {
    const int s0 = 2 * sp, s1 = 2 * sp + 1;
    // Compiler memory barrier: forbids reusing/hoisting whfrag loads across
    // sample iterations (they are otherwise provably loop-invariant).
    asm volatile("" ::: "memory");
    __syncthreads();  // prior iteration's featA reads done
    // Build the two dropout-masked A tiles for this sample pair.
    for (int i = tid; i < 16 * FDIM; i += 256) {
      const int m = i >> 9, k = i & (FDIM - 1);
      const uint32_t base = ((uint32_t)(row0 + m) * (uint32_t)FDIM + (uint32_t)k) * 16u;
      const float     f   = (float)featB[m * PITCH + k] * KEEP_INV;
      const uint32_t h0 = mix32((base + (uint32_t)s0) ^ 0x9e3779b9u);
      const uint32_t h1 = mix32((base + (uint32_t)s1) ^ 0x9e3779b9u);
      featA0[m * PITCH + k] = (_Float16)((h0 < KEEP_THRESH) ? f : 0.0f);
      featA1[m * PITCH + k] = (_Float16)((h1 < KEEP_THRESH) ? f : 0.0f);
    }
    __syncthreads();

    v8f d0[8], d1[8];
#pragma unroll
    for (int nt = 0; nt < 8; ++nt) { d0[nt] = (v8f)0.0f; d1[nt] = (v8f)0.0f; }

    // Rolled K loop: B-fragment addresses depend on ks, so loads cannot be
    // hoisted out of the sample loop (this caused scratch spills before).
#pragma clang loop unroll(disable)
    for (int ks = 0; ks < 16; ++ks) {
      // A fragment (contiguous K, matches B packing): e -> K=16h+e.
      const _Float16* ap0 = &featA0[m16 * PITCH + ks * 32 + h * 16];
      const _Float16* ap1 = &featA1[m16 * PITCH + ks * 32 + h * 16];
      const v16h a0 = __builtin_shufflevector(*(const v8h*)ap0, *(const v8h*)(ap0 + 8),
          0, 1, 2, 3, 4, 5, 6, 7, 8, 9, 10, 11, 12, 13, 14, 15);
      const v16h a1 = __builtin_shufflevector(*(const v8h*)ap1, *(const v8h*)(ap1 + 8),
          0, 1, 2, 3, 4, 5, 6, 7, 8, 9, 10, 11, 12, 13, 14, 15);

      const v16h* bp = &whfrag[((size_t)ks * 64 + (size_t)wv * 8) * 32 + lane];
      // Preload all 8 B fragments into distinct buffers: one big load clause,
      // incremental waits, up to 8 fragments in flight behind the WMMAs.
      v16h bf[8];
#pragma unroll
      for (int nt = 0; nt < 8; ++nt) bf[nt] = bp[nt * 32];
#pragma unroll
      for (int nt = 0; nt < 8; ++nt) {
        d0[nt] = __builtin_amdgcn_wmma_f32_16x16x32_f16(
            false, a0, false, bf[nt], (short)0, d0[nt], false, false);
        d1[nt] = __builtin_amdgcn_wmma_f32_16x16x32_f16(
            false, a1, false, bf[nt], (short)0, d1[nt], false, false);
      }
    }

    softmax_accum(d0, bb, accS, accS2, lane, wv, partial, rowval);
    softmax_accum(d1, bb, accS, accS2, lane, wv, partial, rowval);
  }

  // total_var(b) = sum_c (Sum p^2 - (Sum p)^2 / S) / (S-1)
  float pv[8];
#pragma unroll
  for (int r = 0; r < 8; ++r) pv[r] = 0.0f;
#pragma unroll
  for (int nt = 0; nt < 8; ++nt) {
#pragma unroll
    for (int r = 0; r < 8; ++r) {
      const float sp = accS[nt][r];
      pv[r] += accS2[nt][r] - sp * sp * (1.0f / (float)S_N);
    }
  }
  float tv[8];
  row_reduce<false>(pv, tv, lane, wv, partial, rowval);
  if (tid < 16) {
    out[row0 + tid] = logf(rowval[tid] * (1.0f / (float)(S_N - 1)) + 1e-12f);
  }
}

extern "C" void kernel_launch(void* const* d_in, const int* in_sizes, int n_in,
                              void* d_out, int out_size, void* d_ws, size_t ws_size,
                              hipStream_t stream) {
  (void)in_sizes; (void)n_in; (void)out_size; (void)ws_size;
  const float* feat = (const float*)d_in[0];   // (32768, 512)
  // d_in[1] = logits, unused by reference
  const float* W    = (const float*)d_in[2];   // (512, 1000)
  const float* bias = (const float*)d_in[3];   // (1000,)
  float* out = (float*)d_out;                  // (32768, 1)

  _Float16* whfrag = (_Float16*)d_ws;                         // 524288 halves = 1 MB
  float* bpad = (float*)((char*)d_ws + (size_t)524288 * 2);   // 1024 floats

  pack_w_frag<<<2048, 256, 0, stream>>>(W, whfrag);
  pack_bias<<<4, 256, 0, stream>>>(bias, bpad);
  mc_dropout_var<<<BDIM / 16, 256, 0, stream>>>(feat, (const v16h*)whfrag, bpad, out);
}
